// GroupedQueryAttention_69544110457458
// MI455X (gfx1250) — matrix operand order repaired and checked by reading, not compiled, not run on hardware
//
#include <hip/hip_runtime.h>

#define DH   128
#define TT   1024
#define MM   2048
#define GG   4
#define HH   4
#define BB   2
#define GH   16      // G*H
#define BT   2048    // B*T
#define KOUT 2048    // G*H*D

typedef _Float16 v16h __attribute__((ext_vector_type(16)));
typedef _Float16 v8h  __attribute__((ext_vector_type(8)));
typedef float    v8f  __attribute__((ext_vector_type(8)));

union AFrag { v16h v; v8h h[2]; };

// A-operand (16x32 f16, M x K) from a row-major [rows][ld] f16 matrix.
// ISA layout: lanes 0-15: M=lane, K=0..7 (h[0]) and 16..23 (h[1]);
//             lanes 16-31: M=lane-16, K=8..15 and 24..31.
__device__ inline v16h load_a_f16(const _Float16* __restrict__ base, int ld,
                                  int row_base, int k_base) {
  const int lane = threadIdx.x & 31;
  AFrag u;
  const _Float16* p = base + (size_t)(row_base + (lane & 15)) * ld
                           + k_base + ((lane >> 4) << 3);
  u.h[0] = *(const v8h*)(p);
  u.h[1] = *(const v8h*)(p + 16);
  return u.v;
}

// B-operand (32x16 f16, K x N) from an N-major [N][ld] f16 matrix.
// ISA layout: lanes 0-15: N=lane, K=0..7 and 16..23; lanes 16-31: N=lane-16, K=8..15 and 24..31.
__device__ inline v16h load_b_nmajor(const _Float16* __restrict__ base, int ld,
                                     int n_base, int k_base) {
  const int lane = threadIdx.x & 31;
  AFrag u;
  const _Float16* p = base + (size_t)(n_base + (lane & 15)) * ld
                           + k_base + ((lane >> 4) << 3);
  u.h[0] = *(const v8h*)(p);
  u.h[1] = *(const v8h*)(p + 16);
  return u.v;
}

// B-operand from an LDS-staged n-major [16 n][64 k] f16 sub-tile
// (tile pre-offset by nt*16*64), k sub-chunk kc in {0,32}.
__device__ inline v16h load_b_lds64(const _Float16* tile, int kc) {
  const int lane = threadIdx.x & 31;
  AFrag u;
  const _Float16* p = tile + (lane & 15) * 64 + kc + ((lane >> 4) << 3);
  u.h[0] = *(const v8h*)(p);
  u.h[1] = *(const v8h*)(p + 16);
  return u.v;
}

__device__ inline v8f wmma16x16x32(v16h a, v16h b, v8f c) {
  return __builtin_amdgcn_wmma_f32_16x16x32_f16(false, a, false, b, (short)0, c,
                                                false, false);
}

// One async 16B-per-lane global->LDS chunk (512B per wave instruction, ASYNCcnt).
__device__ inline void async_chunk(const _Float16* g, uint32_t lds_byte_off) {
  asm volatile("global_load_async_to_lds_b128 %0, %1, off"
               :: "v"(lds_byte_off), "v"(g) : "memory");
}

__device__ inline void wait_async0() {
  asm volatile("s_wait_asynccnt 0" ::: "memory");
}

// Stage a [128 n][64 k] f16 B-tile (16KB) into LDS: 1024 16B chunks over 256 threads.
__device__ inline void issue_btile64(const _Float16* __restrict__ src, int ld,
                                     int n_base, int kb, _Float16* tile, int tid) {
#pragma unroll
  for (int r = 0; r < 4; ++r) {
    const int c   = tid + r * 256;
    const int row = c >> 3;
    const int ch  = (c & 7) << 3;   // half-word offset within row: 0..56
    const _Float16* g = src + (size_t)(n_base + row) * ld + kb + ch;
    async_chunk(g, (uint32_t)(uintptr_t)(tile + row * 64 + ch));
  }
}

// ---------------- Stage 1: convert / transpose to f16 ----------------

__global__ void cvt_f32_to_f16(const float* __restrict__ in,
                               _Float16* __restrict__ out, int n, float scale) {
  int i = blockIdx.x * blockDim.x + threadIdx.x;
  if (i < n) out[i] = (_Float16)(in[i] * scale);
}

// in: [batch][R][C] row-major -> out: [batch][C][R]  (batch = blockIdx.y)
__global__ void transpose_cvt(const float* __restrict__ in,
                              _Float16* __restrict__ out,
                              int R, int C, float scale) {
  const size_t bofs = (size_t)blockIdx.y * R * C;
  int i = blockIdx.x * blockDim.x + threadIdx.x;   // out-plane index c*R + r
  if (i < R * C) {
    int c = i / R, r = i - c * R;
    out[bofs + i] = (_Float16)(in[bofs + (size_t)r * C + c] * scale);
  }
}

// ---------------- Stage 2: QKV projection + in-register RoPE ----------------
// grid = (BT/128, 24); block = 256 (8 waves = 2 waves/SIMD). Wave: 16 rows x 128 cols.
// B staged 64-deep to LDS (2 x 16KB, async double-buffered); A double-buffered in regs.

__global__ void __launch_bounds__(256, 2)
qkv_rope_kernel(const _Float16* __restrict__ xh,
                const _Float16* __restrict__ wqT,
                const _Float16* __restrict__ wkT,
                const _Float16* __restrict__ wvT,
                _Float16* __restrict__ qo,
                _Float16* __restrict__ ko,
                _Float16* __restrict__ vTo) {
  __shared__ __align__(16) _Float16 bs[2][DH * 64];   // 2 x 16KB
  const int tid     = threadIdx.x;
  const int lane    = tid & 31;
  const int wave    = tid >> 5;
  const int rowBase = blockIdx.x * 128 + wave * 16;
  const int mat     = blockIdx.y;          // 0..15 q heads, 16..19 k, 20..23 v

  const _Float16* w;
  if (mat < GH)            w = wqT + (size_t)mat * DH * MM;
  else if (mat < GH + GG)  w = wkT + (size_t)(mat - GH) * DH * MM;
  else                     w = wvT + (size_t)(mat - GH - GG) * DH * MM;

  issue_btile64(w, MM, 0, 0, &bs[0][0], tid);
  v16h a0 = load_a_f16(xh, MM, rowBase, 0);
  v16h a1 = load_a_f16(xh, MM, rowBase, 32);

  v8f acc[8] = {};
  for (int st = 0; st < 32; ++st) {         // 32 stages x 64-deep K
    wait_async0();          // my async chunks for current buffer done
    __syncthreads();        // everyone's chunks done; prev readers finished
    if (st + 1 < 32)
      issue_btile64(w, MM, 0, (st + 1) * 64, &bs[(st + 1) & 1][0], tid);
    v16h a0n = a0, a1n = a1;
    if (st + 1 < 32) {
      a0n = load_a_f16(xh, MM, rowBase, (st + 1) * 64);
      a1n = load_a_f16(xh, MM, rowBase, (st + 1) * 64 + 32);
    }
    const _Float16* bt = &bs[st & 1][0];
#pragma unroll
    for (int nt = 0; nt < 8; ++nt)
      acc[nt] = wmma16x16x32(a0, load_b_lds64(bt + nt * 16 * 64, 0), acc[nt]);
#pragma unroll
    for (int nt = 0; nt < 8; ++nt)
      acc[nt] = wmma16x16x32(a1, load_b_lds64(bt + nt * 16 * 64, 32), acc[nt]);
    a0 = a0n; a1 = a1n;
  }

  const int bidx  = rowBase >> 10;   // batch
  const int tBase = rowBase & 1023;  // position within sequence
  const int half  = lane >> 4;
  const int nlane = lane & 15;

  if (mat < GH + GG) {
    // split-half RoPE: even = d in [0,64) (tiles 0..3), odd = d+64 (tiles 4..7)
#pragma unroll
    for (int nt = 0; nt < 4; ++nt) {
      const float d    = (float)(nt * 16 + nlane);
      const float freq = __powf(10000.0f, -d * (1.0f / 64.0f));
#pragma unroll
      for (int j = 0; j < 8; ++j) {
        float t = (float)(tBase + j + 8 * half);
        float sn, cs;
        __sincosf(t * freq, &sn, &cs);
        float e = acc[nt][j], o = acc[nt + 4][j];
        acc[nt][j]     = e * cs - o * sn;
        acc[nt + 4][j] = e * sn + o * cs;
      }
    }
  }

  if (mat < GH) {                                    // q: [B][GH][T][D]
    _Float16* dst = qo + (size_t)(bidx * GH + mat) * TT * DH;
#pragma unroll
    for (int nt = 0; nt < 8; ++nt)
#pragma unroll
      for (int j = 0; j < 8; ++j)
        dst[(size_t)(tBase + j + 8 * half) * DH + nt * 16 + nlane] =
            (_Float16)acc[nt][j];
  } else if (mat < GH + GG) {                        // k: [B][G][T][D]
    _Float16* dst = ko + (size_t)(bidx * GG + (mat - GH)) * TT * DH;
#pragma unroll
    for (int nt = 0; nt < 8; ++nt)
#pragma unroll
      for (int j = 0; j < 8; ++j)
        dst[(size_t)(tBase + j + 8 * half) * DH + nt * 16 + nlane] =
            (_Float16)acc[nt][j];
  } else {                                           // v transposed: [B][G][D][T]
    _Float16* dst = vTo + (size_t)(bidx * GG + (mat - GH - GG)) * DH * TT;
#pragma unroll
    for (int nt = 0; nt < 8; ++nt)
#pragma unroll
      for (int j = 0; j < 8; ++j)
        dst[(size_t)(nt * 16 + nlane) * TT + tBase + j + 8 * half] =
            (_Float16)acc[nt][j];
  }
}

// ---------------- Stage 3: flash attention per (b,g,h) ----------------
// grid = (T/128, B*G*H); block = 256 (8 waves). Wave owns 16 query rows.

__global__ void __launch_bounds__(256, 2)
attn_kernel(const _Float16* __restrict__ q,
            const _Float16* __restrict__ k,
            const _Float16* __restrict__ vT,
            _Float16* __restrict__ o) {
  __shared__ __align__(16) _Float16 lds_p[8][16 * 32];   // per-wave P tile
  const int lane = threadIdx.x & 31;
  const int wave = threadIdx.x >> 5;
  const int hb   = blockIdx.y;
  const int b    = hb >> 4;
  const int gh   = hb & 15;
  const int g    = gh >> 2;
  const int q0   = blockIdx.x * 128 + wave * 16;

  const _Float16* qp = q  + (size_t)(b * GH + gh) * TT * DH;
  const _Float16* kp = k  + (size_t)(b * GG + g)  * TT * DH;
  const _Float16* vp = vT + (size_t)(b * GG + g)  * DH * TT;
  _Float16*       op = o  + (size_t)b * TT * KOUT + (size_t)gh * DH;

  v16h aq[4];
#pragma unroll
  for (int kc = 0; kc < 4; ++kc) aq[kc] = load_a_f16(qp, DH, q0, kc * 32);

  v8f oacc[8] = {};
  float mrow[8], lrow[8];
#pragma unroll
  for (int j = 0; j < 8; ++j) { mrow[j] = -3.0e38f; lrow[j] = 0.0f; }

  const int half  = lane >> 4;
  const int nlane = lane & 15;
  const int jbMax = (q0 + 15) >> 5;

  for (int jb = 0; jb <= jbMax; ++jb) {
    const int j0 = jb * 32;

    // ---- S = q k^T over D=128: load all 8 K-frags, then 8 WMMAs ----
    v16h kf[8];
#pragma unroll
    for (int kc = 0; kc < 4; ++kc) {
      kf[kc]     = load_b_nmajor(kp, DH, j0,      kc * 32);
      kf[kc + 4] = load_b_nmajor(kp, DH, j0 + 16, kc * 32);
    }
    v8f s0 = {}, s1 = {};
#pragma unroll
    for (int kc = 0; kc < 4; ++kc) {
      s0 = wmma16x16x32(aq[kc], kf[kc],     s0);
      s1 = wmma16x16x32(aq[kc], kf[kc + 4], s1);
    }

    // ---- issue V-frag loads early; softmax VALU hides their latency ----
    v16h vf[8];
#pragma unroll
    for (int dt = 0; dt < 8; ++dt) vf[dt] = load_b_nmajor(vp, TT, dt * 16, j0);

    // causal mask (s -= 1e30 where col > row, as in the reference)
#pragma unroll
    for (int j = 0; j < 8; ++j) {
      const int row = q0 + j + 8 * half;
      if (j0 + nlane > row)      s0[j] -= 1.0e30f;
      if (j0 + 16 + nlane > row) s1[j] -= 1.0e30f;
    }
    // online softmax: rows live across the 16 lanes of each half-wave
#pragma unroll
    for (int j = 0; j < 8; ++j) {
      float mb = fmaxf(s0[j], s1[j]);
      mb = fmaxf(mb, __shfl_xor(mb, 1));
      mb = fmaxf(mb, __shfl_xor(mb, 2));
      mb = fmaxf(mb, __shfl_xor(mb, 4));
      mb = fmaxf(mb, __shfl_xor(mb, 8));
      float mnew  = fmaxf(mrow[j], mb);
      float alpha = __expf(mrow[j] - mnew);
      mrow[j] = mnew;
      float p0 = __expf(s0[j] - mnew);
      float p1 = __expf(s1[j] - mnew);
      s0[j] = p0; s1[j] = p1;
      float rs = p0 + p1;
      rs += __shfl_xor(rs, 1);
      rs += __shfl_xor(rs, 2);
      rs += __shfl_xor(rs, 4);
      rs += __shfl_xor(rs, 8);
      lrow[j] = lrow[j] * alpha + rs;
#pragma unroll
      for (int dt = 0; dt < 8; ++dt) oacc[dt][j] *= alpha;
    }
    // C-layout P -> LDS (row-major 16x32 f16) -> reload in A-operand layout
    _Float16* pl = &lds_p[wave][0];
#pragma unroll
    for (int j = 0; j < 8; ++j) {
      const int r = j + 8 * half;
      pl[r * 32 + nlane]      = (_Float16)s0[j];
      pl[r * 32 + 16 + nlane] = (_Float16)s1[j];
    }
    __builtin_amdgcn_wave_barrier();   // LDS ops are in-order within a wave
    AFrag pa;
    {
      const _Float16* pp = pl + nlane * 32 + half * 8;
      pa.h[0] = *(const v8h*)(pp);
      pa.h[1] = *(const v8h*)(pp + 16);
    }
    __builtin_amdgcn_wave_barrier();
#pragma unroll
    for (int dt = 0; dt < 8; ++dt)
      oacc[dt] = wmma16x16x32(pa.v, vf[dt], oacc[dt]);
  }

#pragma unroll
  for (int j = 0; j < 8; ++j) {
    const float inv = 1.0f / lrow[j];
    const int t = q0 + j + 8 * half;
#pragma unroll
    for (int dt = 0; dt < 8; ++dt)
      op[(size_t)t * KOUT + dt * 16 + nlane] = (_Float16)(oacc[dt][j] * inv);
  }
}

// ---------------- Stage 4: output projection [2048x2048x2048] ----------------
// grid = (BT/128, MM/128); block = 256. Wave: 16 rows x 128 cols.
// B staged 64-deep to LDS (2 x 16KB, async double-buffered).

__global__ void __launch_bounds__(256, 2)
out_proj_kernel(const _Float16* __restrict__ o,
                const _Float16* __restrict__ woT,
                float* __restrict__ out) {
  __shared__ __align__(16) _Float16 bs[2][128 * 64];   // 2 x 16KB
  const int tid     = threadIdx.x;
  const int lane    = tid & 31;
  const int wave    = tid >> 5;
  const int rowBase = blockIdx.x * 128 + wave * 16;
  const int colBase = blockIdx.y * 128;

  issue_btile64(woT, KOUT, colBase, 0, &bs[0][0], tid);
  v16h a0 = load_a_f16(o, KOUT, rowBase, 0);
  v16h a1 = load_a_f16(o, KOUT, rowBase, 32);

  v8f acc[8] = {};
  for (int st = 0; st < 32; ++st) {
    wait_async0();
    __syncthreads();
    if (st + 1 < 32)
      issue_btile64(woT, KOUT, colBase, (st + 1) * 64, &bs[(st + 1) & 1][0], tid);
    v16h a0n = a0, a1n = a1;
    if (st + 1 < 32) {
      a0n = load_a_f16(o, KOUT, rowBase, (st + 1) * 64);
      a1n = load_a_f16(o, KOUT, rowBase, (st + 1) * 64 + 32);
    }
    const _Float16* bt = &bs[st & 1][0];
#pragma unroll
    for (int nt = 0; nt < 8; ++nt)
      acc[nt] = wmma16x16x32(a0, load_b_lds64(bt + nt * 16 * 64, 0), acc[nt]);
#pragma unroll
    for (int nt = 0; nt < 8; ++nt)
      acc[nt] = wmma16x16x32(a1, load_b_lds64(bt + nt * 16 * 64, 32), acc[nt]);
    a0 = a0n; a1 = a1n;
  }
  const int half = lane >> 4, nlane = lane & 15;
#pragma unroll
  for (int nt = 0; nt < 8; ++nt)
#pragma unroll
    for (int j = 0; j < 8; ++j)
      out[(size_t)(rowBase + j + 8 * half) * MM + colBase + nt * 16 + nlane] =
          acc[nt][j];
}

// ---------------- launch ----------------

extern "C" void kernel_launch(void* const* d_in, const int* in_sizes, int n_in,
                              void* d_out, int out_size, void* d_ws, size_t ws_size,
                              hipStream_t stream) {
  (void)in_sizes; (void)n_in; (void)out_size; (void)ws_size;
  const float* x    = (const float*)d_in[0];
  const float* w_aq = (const float*)d_in[1];
  const float* w_ak = (const float*)d_in[2];
  const float* w_av = (const float*)d_in[3];
  const float* w_ao = (const float*)d_in[4];
  float* out = (float*)d_out;

  char* ws = (char*)d_ws;
  _Float16* xh   = (_Float16*)ws; ws += (size_t)BT * MM * 2;        // x in f16
  _Float16* wqT  = (_Float16*)ws; ws += (size_t)GH * DH * MM * 2;   // [16][D][M]
  _Float16* wkT  = (_Float16*)ws; ws += (size_t)GG * DH * MM * 2;   // [4][D][M]
  _Float16* wvT  = (_Float16*)ws; ws += (size_t)GG * DH * MM * 2;   // [4][D][M]
  _Float16* woT  = (_Float16*)ws; ws += (size_t)MM * KOUT * 2;      // [M][GHD]
  _Float16* qbuf = (_Float16*)ws; ws += (size_t)BB * GH * TT * DH * 2;
  _Float16* kbuf = (_Float16*)ws; ws += (size_t)BB * GG * TT * DH * 2;
  _Float16* vTbf = (_Float16*)ws; ws += (size_t)BB * GG * DH * TT * 2;
  _Float16* obuf = (_Float16*)ws; ws += (size_t)BB * TT * KOUT * 2;

  const float qs = 0.29730177875068026f;  // 128^(-1/4), folded into wq and wk

  cvt_f32_to_f16<<<(BT * MM + 255) / 256, 256, 0, stream>>>(x, xh, BT * MM, 1.0f);
  transpose_cvt<<<dim3((MM * DH + 255) / 256, GH), 256, 0, stream>>>(w_aq, wqT, MM, DH, qs);
  transpose_cvt<<<dim3((MM * DH + 255) / 256, GG), 256, 0, stream>>>(w_ak, wkT, MM, DH, qs);
  transpose_cvt<<<dim3((MM * DH + 255) / 256, GG), 256, 0, stream>>>(w_av, wvT, MM, DH, 1.0f);
  transpose_cvt<<<dim3((KOUT * MM + 255) / 256, 1), 256, 0, stream>>>(w_ao, woT, KOUT, MM, 1.0f);

  qkv_rope_kernel<<<dim3(BT / 128, GH + 2 * GG), 256, 0, stream>>>(
      xh, wqT, wkT, wvT, qbuf, kbuf, vTbf);
  attn_kernel<<<dim3(TT / 128, BB * GH), 256, 0, stream>>>(qbuf, kbuf, vTbf, obuf);
  out_proj_kernel<<<dim3(BT / 128, MM / 128), 256, 0, stream>>>(obuf, woT, out);
}
